// PcConvBp_SGD_77154792505984
// MI455X (gfx1250) — matrix operand order repaired and checked by reading, not compiled, not run on hardware
//
#include <hip/hip_runtime.h>

// ---------------------------------------------------------------------------
// PcConvBp_SGD for MI455X (gfx1250, wave32, WMMA bf16 16x16x32)
//
// B=64, DX=8192, DY=10368, 5 SGD iters -> 10 skinny GEMMs vs W_exp.
// Single bf16 copy of W_exp (170MB) => fits 192MB L2, all 10 passes L2-fed.
//   R-GEMM  (K=DY): Wb rows are K-contiguous -> direct streaming WMMA loop.
//   G-GEMM  (K=DX): needs W transposed -> per-wave LDS transpose stage
//                   (v_perm pack + ds_store_b32 scatter, padded rows, no
//                   barriers needed: per-wave LDS ops are in-order).
// ---------------------------------------------------------------------------

typedef __bf16        v16bf __attribute__((ext_vector_type(16)));
typedef __bf16        v4bf  __attribute__((ext_vector_type(4)));
typedef float         v8f   __attribute__((ext_vector_type(8)));
typedef float         vf4   __attribute__((ext_vector_type(4)));
typedef unsigned int  vu4   __attribute__((ext_vector_type(4)));

#define DX   8192
#define DY   10368
#define NB   64
#define LRF  0.01f

struct PairF { vf4 x, y; };
struct PairU { vu4 x, y; };

__device__ __forceinline__ v16bf pack_bf(vf4 a, vf4 b) {
  PairF p{a, b};
  return __builtin_bit_cast(v16bf, p);
}
__device__ __forceinline__ v16bf pack_bfu(vu4 a, vu4 b) {
  PairU p{a, b};
  return __builtin_bit_cast(v16bf, p);
}

__device__ __forceinline__ v8f wmma_bf16(v16bf a, v16bf b, v8f c) {
  return __builtin_amdgcn_wmma_f32_16x16x32_bf16(
      false, a, false, b, (short)0, c, false, false);
}

// ---------------------------------------------------------------------------
// Prep: W_exp fp32 -> Wb bf16 (pure convert, 4 elems/thread). Zero e2[0..7].
// ---------------------------------------------------------------------------
__global__ __launch_bounds__(256) void k_prep(const float* __restrict__ W,
                                              __bf16* __restrict__ Wb,
                                              float* __restrict__ e2) {
  const size_t i = ((size_t)blockIdx.x * 256 + threadIdx.x) * 4;  // 84934656 total
  vf4 w = *(const vf4*)(W + i);
  v4bf o;
  o[0] = (__bf16)w[0]; o[1] = (__bf16)w[1];
  o[2] = (__bf16)w[2]; o[3] = (__bf16)w[3];
  *(v4bf*)(Wb + i) = o;
  if (blockIdx.x == 0 && threadIdx.x < 8) e2[threadIdx.x] = 0.f;
}

// ---------------------------------------------------------------------------
// Feed-forward: flat_y = pad(relu(conv3x3(x, Wff))) fp32 + bf16 copy yb.
// ---------------------------------------------------------------------------
__global__ __launch_bounds__(256) void k_ff(const float* __restrict__ x,
                                            const float* __restrict__ wff,
                                            float* __restrict__ fy,
                                            __bf16* __restrict__ yb) {
  __shared__ float wsh[32 * 32 * 9];
  for (int i = threadIdx.x; i < 32 * 32 * 9; i += 256) wsh[i] = wff[i];
  __syncthreads();
  const int t = blockIdx.x * 256 + threadIdx.x;      // 64*10368 = 663552
  const int b = t / DY;
  int r = t - b * DY;
  const int c = r / 324; r -= c * 324;
  const int ii = r / 18, jj = r - ii * 18;
  float acc = 0.f;
  if (ii >= 1 && ii <= 16 && jj >= 1 && jj <= 16) {
    const int i0 = ii - 1, j0 = jj - 1;
    const float* xb = x + (size_t)b * (32 * 256);
    const float* wc = wsh + c * (32 * 9);
    for (int cin = 0; cin < 32; ++cin) {
      const float* xc = xb + cin * 256;
      const float* wk = wc + cin * 9;
      #pragma unroll
      for (int kh = 0; kh < 3; ++kh) {
        const int yy = i0 + kh - 1;
        if (yy < 0 || yy > 15) continue;
        #pragma unroll
        for (int kw = 0; kw < 3; ++kw) {
          const int xx = j0 + kw - 1;
          if (xx < 0 || xx > 15) continue;
          acc += xc[yy * 16 + xx] * wk[kh * 3 + kw];
        }
      }
    }
    acc = fmaxf(acc, 0.f);
  }
  fy[t] = acc;
  yb[t] = (__bf16)acc;
}

// ---------------------------------------------------------------------------
// R-GEMM (direct): part[sp][b][n] = sum_k A[b][k]*Bm[n][k]
// A bf16 [64 x K] row-major, Bm bf16 [N x K] row-major (K-contiguous rows).
// One wave per (n-tile, K-split); 4 accumulators cover batch M=64.
// ---------------------------------------------------------------------------
__global__ __launch_bounds__(32) void k_gemm(const __bf16* __restrict__ A,
                                             const __bf16* __restrict__ Bm,
                                             float* __restrict__ part,
                                             int K, int N) {
  const int nt = blockIdx.x;
  const int sp = blockIdx.y;          // K-split 0..3
  const int KS = K >> 2;
  const int lane = threadIdx.x;
  const int lo = lane & 15, hi = lane >> 4;
  const int n0 = nt * 16;

  const __bf16* bp = Bm + (size_t)(n0 + lo) * K + sp * KS + hi * 8;
  const __bf16* ap = A + (size_t)lo * K + sp * KS + hi * 8;
  const size_t sA = (size_t)16 * K;

  v8f acc0 = {}, acc1 = {}, acc2 = {}, acc3 = {};
  for (int k = 0; k < KS; k += 32) {
    v16bf bm = pack_bf(*(const vf4*)bp, *(const vf4*)(bp + 16));
    v16bf a0 = pack_bf(*(const vf4*)ap,            *(const vf4*)(ap + 16));
    v16bf a1 = pack_bf(*(const vf4*)(ap + sA),     *(const vf4*)(ap + sA + 16));
    v16bf a2 = pack_bf(*(const vf4*)(ap + 2 * sA), *(const vf4*)(ap + 2 * sA + 16));
    v16bf a3 = pack_bf(*(const vf4*)(ap + 3 * sA), *(const vf4*)(ap + 3 * sA + 16));
    acc0 = wmma_bf16(a0, bm, acc0);
    acc1 = wmma_bf16(a1, bm, acc1);
    acc2 = wmma_bf16(a2, bm, acc2);
    acc3 = wmma_bf16(a3, bm, acc3);
    bp += 32;
    ap += 32;
#if defined(__gfx1250__)
    __builtin_prefetch(bp + 512, 0, 1);   // global_prefetch_b8 on W stream
#endif
  }
  float* o = part + ((size_t)sp * NB + hi * 8) * N + n0 + lo;
  #pragma unroll
  for (int v = 0; v < 8; ++v) {
    o[(size_t)(v)      * N] = acc0[v];
    o[(size_t)(v + 16) * N] = acc1[v];
    o[(size_t)(v + 32) * N] = acc2[v];
    o[(size_t)(v + 48) * N] = acc3[v];
  }
}

// ---------------------------------------------------------------------------
// G-GEMM (LDS-transposed W): part[sp][b][n] = sum_k A[b][k]*Wb[k][n]
// Wb read in its native [k][n] order (n-contiguous, fully coalesced, stays
// L2-resident); each wave transposes a 32k x 16n tile into LDS per K-step:
//   lane (p=lane%16, h=lane/16) loads rows k0+2p / k0+2p+1 cols h*8..h*8+7,
//   v_perm pairs {k+1,k} into dwords, 8x ds_store_b32 scatter into rows of a
//   padded [16][RS=20 dwords] tile; B operand read back as ds_load_b128 x2.
// Per-wave LDS ops are in-order => single buffer, no barriers.
// ---------------------------------------------------------------------------
__global__ __launch_bounds__(32) void k_gemm_t(const __bf16* __restrict__ A,
                                               const __bf16* __restrict__ Wb,
                                               float* __restrict__ part) {
  constexpr int K = DX, N = DY, RS = 20;       // RS dwords per padded LDS row
  __shared__ unsigned int ldsT[16 * RS];       // 1280 B, wave-private
  const int nt = blockIdx.x;                   // 648 n-tiles
  const int sp = blockIdx.y;                   // K-split 0..3
  const int KS = K >> 2;                       // 2048
  const int lane = threadIdx.x;
  const int lo = lane & 15, hi = lane >> 4;
  const int n0 = nt * 16;

  const __bf16* gp = Wb + (size_t)(sp * KS + 2 * lo) * N + n0 + hi * 8;
  const __bf16* ap = A + (size_t)lo * K + sp * KS + hi * 8;
  const size_t sA = (size_t)16 * K;

  unsigned int* st = &ldsT[(hi * 8) * RS + lo];          // + j*RS per n-row
  const unsigned int* ldb = &ldsT[lo * RS + hi * 4];     // B row lo, k 8h..8h+7 / 16+8h..23+8h

  v8f acc0 = {}, acc1 = {}, acc2 = {}, acc3 = {};
  for (int k = 0; k < KS; k += 32) {
    vu4 ra = *(const vu4*)gp;                  // row k0+2p : 8 bf16 (n-chunk)
    vu4 rb = *(const vu4*)(gp + N);            // row k0+2p+1
    gp += (size_t)32 * N;
    #pragma unroll
    for (int j = 0; j < 4; ++j) {
      // dword = {W[k0+2p+1][n] , W[k0+2p][n]} for n = n0+h*8+2j (+1)
      unsigned d0 = __builtin_amdgcn_perm(rb[j], ra[j], 0x05040100u);
      unsigned d1 = __builtin_amdgcn_perm(rb[j], ra[j], 0x07060302u);
      st[(2 * j)     * RS] = d0;
      st[(2 * j + 1) * RS] = d1;
    }
    // B operand: lane lo = column n, k 8h..8h+7 then 16+8h..16+8h+7
    v16bf bm = pack_bfu(*(const vu4*)ldb, *(const vu4*)(ldb + 8));
    v16bf a0 = pack_bf(*(const vf4*)ap,            *(const vf4*)(ap + 16));
    v16bf a1 = pack_bf(*(const vf4*)(ap + sA),     *(const vf4*)(ap + sA + 16));
    v16bf a2 = pack_bf(*(const vf4*)(ap + 2 * sA), *(const vf4*)(ap + 2 * sA + 16));
    v16bf a3 = pack_bf(*(const vf4*)(ap + 3 * sA), *(const vf4*)(ap + 3 * sA + 16));
    acc0 = wmma_bf16(a0, bm, acc0);
    acc1 = wmma_bf16(a1, bm, acc1);
    acc2 = wmma_bf16(a2, bm, acc2);
    acc3 = wmma_bf16(a3, bm, acc3);
    ap += 32;
  }
  float* o = part + ((size_t)sp * NB + hi * 8) * N + n0 + lo;
  #pragma unroll
  for (int v = 0; v < 8; ++v) {
    o[(size_t)(v)      * N] = acc0[v];
    o[(size_t)(v + 16) * N] = acc1[v];
    o[(size_t)(v + 32) * N] = acc2[v];
    o[(size_t)(v + 48) * N] = acc3[v];
  }
}

// ---------------------------------------------------------------------------
// r = flat_x - sum(splits); bf16 r + per-block ||r||^2 partial.
// ---------------------------------------------------------------------------
__global__ __launch_bounds__(256) void k_cast_r(const float* __restrict__ fx,
                                                const float* __restrict__ part,
                                                __bf16* __restrict__ rb,
                                                float* __restrict__ bpart) {
  __shared__ float sm[256];
  const int t = blockIdx.x * 256 + threadIdx.x;   // 524288
  const int n = NB * DX;
  float r = fx[t] - (part[t] + part[t + n] + part[t + 2 * n] + part[t + 3 * n]);
  rb[t] = (__bf16)r;
  sm[threadIdx.x] = r * r;
  __syncthreads();
  for (int o = 128; o > 0; o >>= 1) {
    if (threadIdx.x < o) sm[threadIdx.x] += sm[threadIdx.x + o];
    __syncthreads();
  }
  if (threadIdx.x == 0) bpart[blockIdx.x] = sm[0];
}

// Deterministic fixed-order reduction of 2048 block partials -> e2[iter].
__global__ __launch_bounds__(256) void k_norm(const float* __restrict__ bpart,
                                              float* __restrict__ e2, int iter) {
  __shared__ float sm[256];
  float s = 0.f;
  for (int i = threadIdx.x; i < 2048; i += 256) s += bpart[i];
  sm[threadIdx.x] = s;
  __syncthreads();
  for (int o = 128; o > 0; o >>= 1) {
    if (threadIdx.x < o) sm[threadIdx.x] += sm[threadIdx.x + o];
    __syncthreads();
  }
  if (threadIdx.x == 0) e2[iter] = sm[0];
}

// flat_y += (LR/||r||) * (r @ W_exp); refresh bf16 copy for next R-GEMM.
__global__ __launch_bounds__(256) void k_update(float* __restrict__ fy,
                                                __bf16* __restrict__ yb,
                                                const float* __restrict__ gpart,
                                                const float* __restrict__ e2,
                                                int iter) {
  const int t = blockIdx.x * 256 + threadIdx.x;   // 663552
  const int n = NB * DY;
  float g = gpart[t] + gpart[t + n] + gpart[t + 2 * n] + gpart[t + 3 * n];
  float scale = LRF / sqrtf(e2[iter]);
  float v = fy[t] + scale * g;
  fy[t] = v;
  yb[t] = (__bf16)v;
}

// out = crop(flat_y) + conv1x1(x, Wbyp)
__global__ __launch_bounds__(256) void k_final(const float* __restrict__ x,
                                               const float* __restrict__ wbyp,
                                               const float* __restrict__ fy,
                                               float* __restrict__ out) {
  __shared__ float wsh[1024];
  for (int i = threadIdx.x; i < 1024; i += 256) wsh[i] = wbyp[i];
  __syncthreads();
  const int t = blockIdx.x * 256 + threadIdx.x;   // 524288
  const int b = t >> 13;
  const int r = t & 8191;
  const int c = r >> 8;
  const int p = r & 255;
  const int i = p >> 4, j = p & 15;
  float opt = fy[(size_t)b * DY + c * 324 + (i + 1) * 18 + (j + 1)];
  const float* xb = x + (size_t)b * 8192 + p;
  const float* wc = wsh + c * 32;
  float acc = 0.f;
  #pragma unroll 8
  for (int cin = 0; cin < 32; ++cin) acc += xb[cin * 256] * wc[cin];
  out[t] = opt + acc;
}

// ---------------------------------------------------------------------------
extern "C" void kernel_launch(void* const* d_in, const int* in_sizes, int n_in,
                              void* d_out, int out_size, void* d_ws, size_t ws_size,
                              hipStream_t stream) {
  (void)in_sizes; (void)n_in; (void)out_size; (void)ws_size;
  const float* x    = (const float*)d_in[0];   // [64,32,16,16] == flat_x
  const float* wff  = (const float*)d_in[1];   // [32,32,3,3]
  const float* wbyp = (const float*)d_in[2];   // [32,32,1,1]
  const float* wexp = (const float*)d_in[3];   // [8192,10368]
  float* out = (float*)d_out;

  char* w = (char*)d_ws;
  size_t off = 0;
  auto carve = [&](size_t bytes) -> void* {
    void* p = w + off;
    off = (off + bytes + 255) & ~(size_t)255;
    return p;
  };
  __bf16* Wb    = (__bf16*)carve((size_t)DX * DY * 2);       // 170 MB (L2-resident)
  float*  fy    = (float*) carve((size_t)NB * DY * 4);
  __bf16* yb    = (__bf16*)carve((size_t)NB * DY * 2);
  __bf16* rb    = (__bf16*)carve((size_t)NB * DX * 2);
  float*  rpart = (float*) carve((size_t)4 * NB * DX * 4);
  float*  gpart = (float*) carve((size_t)4 * NB * DY * 4);
  float*  bpart = (float*) carve(2048 * 4);
  float*  e2    = (float*) carve(64);

  k_prep<<<(DX * DY) / 1024, 256, 0, stream>>>(wexp, Wb, e2);
  k_ff<<<(NB * DY) / 256, 256, 0, stream>>>(x, wff, fy, yb);
  for (int it = 0; it < 5; ++it) {
    k_gemm<<<dim3(DX / 16, 4), 32, 0, stream>>>(yb, Wb, rpart, DY, DX);   // r-product
    k_cast_r<<<(NB * DX) / 256, 256, 0, stream>>>(x, rpart, rb, bpart);
    k_norm<<<1, 256, 0, stream>>>(bpart, e2, it);
    k_gemm_t<<<dim3(DY / 16, 4), 32, 0, stream>>>(rb, Wb, gpart);         // gradient
    k_update<<<(NB * DY) / 256, 256, 0, stream>>>(fy, yb, gpart, e2, it);
  }
  k_final<<<(NB * DX) / 256, 256, 0, stream>>>(x, wbyp, fy, out);
}
